// NexusKAN_14508399526592
// MI455X (gfx1250) — matrix operand order repaired and checked by reading, not compiled, not run on hardware
//
#include <hip/hip_runtime.h>
#include <hip/hip_bf16.h>
#include <stdint.h>

#define KAN_ASYNC_LDS 1   // use global_load_async_to_lds_b128 for weight tiles

typedef __attribute__((ext_vector_type(16))) __bf16 v16bf;
typedef __attribute__((ext_vector_type(8)))  __bf16 v8bf;
typedef __attribute__((ext_vector_type(8)))  float  v8f;

// Problem constants (from reference)
constexpr int DDIM  = 512;
constexpr int BATCH = 8192;
constexpr int NB9   = 9;              // 8 spline bases + 1 silu channel
constexpr int KTOT  = DDIM * NB9;     // 4608

// Tiling
constexpr int BM = 128;
constexpr int BN = 128;
constexpr int DI = 32;                // input dims per K-chunk
constexpr int KC = DI * NB9;          // 288 = 9 WMMA K-steps of 32
constexpr int AS_STRIDE = KC + 8;     // 296 halves (592B, 16B aligned, bank-skewed)
constexpr int WS_STRIDE = BN + 8;     // 136 halves (272B, 16B aligned)
constexpr int THREADS = 256;          // 8 waves (wave32)

// ---- pack Wa[k=i*9+c][o]: coefs*scale_sp (c<8) or scale_base (c==8), bf16 ----
__global__ void kan_pack_kernel(const float* __restrict__ coefs,
                                const float* __restrict__ scale_base,
                                const float* __restrict__ scale_sp,
                                __bf16* __restrict__ Wa) {
    const int idx = blockIdx.x * blockDim.x + threadIdx.x;   // < KTOT*DDIM
    const int o    = idx & (DDIM - 1);
    const int kidx = idx >> 9;
    const int i    = kidx / NB9;
    const int c    = kidx - i * NB9;
    const int io   = i * DDIM + o;
    float v = (c < 8) ? coefs[io * 8 + c] * scale_sp[io] : scale_base[io];
    Wa[idx] = (__bf16)v;
}

// ---- fused layer: activations -> bf16 WMMA GEMM -> folded affine epilogue ----
__global__ __launch_bounds__(THREADS) void kan_layer_kernel(
    const float* __restrict__ xin,        // [BATCH, DDIM] fp32
    const __bf16* __restrict__ Wa,        // [KTOT, DDIM] bf16
    const float* __restrict__ p_ss, const float* __restrict__ p_sb,
    const float* __restrict__ p_ns, const float* __restrict__ p_nb,
    float* __restrict__ xout)             // [BATCH, DDIM] fp32
{
    __shared__ __bf16 As[BM * AS_STRIDE];   // 75,776 B
    __shared__ __bf16 Ws[KC * WS_STRIDE];   // 78,336 B

    const int tid  = threadIdx.x;
    const int lane = tid & 31;
    const int wid  = tid >> 5;
    const int wm   = wid & 1;      // 2 waves along M (64 rows each)
    const int wn   = wid >> 1;     // 4 waves along N (32 cols each)
    const int m15  = lane & 15;
    const int hi   = lane >> 4;

    const int blockN = blockIdx.x * BN;
    const int blockM = blockIdx.y * BM;

    v8f acc[4][2] = {};

#pragma unroll 1
    for (int dim0 = 0; dim0 < DDIM; dim0 += DI) {
        // ---- stage W tile (KC x BN bf16) via async global->LDS copies ----
        const __bf16* wsrc = Wa + (size_t)(dim0 * NB9) * DDIM + blockN;
#pragma unroll
        for (int j = 0; j < (KC * BN / 8) / THREADS; ++j) {   // 18 per thread
            const int seg = tid + THREADS * j;
            const int row = seg >> 4;          // 16 segs of 8 halves per row
            const int c8  = (seg & 15) * 8;
#if KAN_ASYNC_LDS
            const unsigned lofs = (unsigned)(uintptr_t)&Ws[row * WS_STRIDE + c8];
            const unsigned long long ga =
                (unsigned long long)(uintptr_t)(wsrc + (size_t)row * DDIM + c8);
            asm volatile("global_load_async_to_lds_b128 %0, %1, off"
                         :: "v"(lofs), "v"(ga) : "memory");
#else
            v8bf v = *(const v8bf*)(wsrc + (size_t)row * DDIM + c8);
            *(v8bf*)&Ws[row * WS_STRIDE + c8] = v;
#endif
        }

        // ---- stage A: activations for BM x DI inputs -> bf16 in LDS ----
        // uniform grid t_j = -1 + 0.4*(j-3); cubic B-spline via direct 4-segment
        // evaluation (only bases m-3..m are nonzero for x in [t_m, t_m+1))
#pragma unroll
        for (int j = 0; j < (BM * DI) / THREADS; ++j) {       // 16 per thread
            const int idx = tid + THREADS * j;
            const int bl  = idx >> 5;
            const int il  = idx & (DI - 1);
            const float xv = xin[(size_t)(blockM + bl) * DDIM + dim0 + il];
            __bf16* dst = &As[bl * AS_STRIDE + il * NB9];

            const __bf16 z = (__bf16)0.0f;
#pragma unroll
            for (int c = 0; c < 8; ++c) dst[c] = z;
            dst[8] = (__bf16)(xv / (1.0f + __expf(-xv)));     // silu channel

            const float fi = (xv + 2.2f) * 2.5f;              // (x - t0)/h
            const float fm = floorf(fi);
            const int   m  = (int)fm;
            if (m >= 0 && m <= 10) {
                const float u  = fi - fm;
                const float u2 = u * u;
                const float u3 = u2 * u;
                const float omu = 1.0f - u;
                const float b0 = (omu * omu * omu) * (1.0f / 6.0f);
                const float b1 = (3.0f * u3 - 6.0f * u2 + 4.0f) * (1.0f / 6.0f);
                const float b2 = (-3.0f * u3 + 3.0f * u2 + 3.0f * u + 1.0f) * (1.0f / 6.0f);
                const float b3 = u3 * (1.0f / 6.0f);
                const int j0 = m - 3;
                if (j0     >= 0 && j0     < 8) dst[j0]     = (__bf16)b0;
                if (j0 + 1 >= 0 && j0 + 1 < 8) dst[j0 + 1] = (__bf16)b1;
                if (j0 + 2 >= 0 && j0 + 2 < 8) dst[j0 + 2] = (__bf16)b2;
                if (j0 + 3 >= 0 && j0 + 3 < 8) dst[j0 + 3] = (__bf16)b3;
            }
        }

        // prefetch next Wa tile into cache (global_prefetch_b8)
        if (dim0 + DI < DDIM) {
            const __bf16* pf = Wa + (size_t)((dim0 + DI) * NB9) * DDIM + blockN;
            __builtin_prefetch(pf + (size_t)tid * DDIM, 0, 1);
        }

#if KAN_ASYNC_LDS
#if __has_builtin(__builtin_amdgcn_s_wait_asynccnt)
        __builtin_amdgcn_s_wait_asynccnt(0);
#else
        asm volatile("s_wait_asynccnt 0" ::: "memory");
#endif
#endif
        __syncthreads();

        // ---- 9 WMMA K-steps over this chunk ----
#pragma unroll
        for (int kb = 0; kb < KC / 32; ++kb) {
            v16bf afrag[4];
#pragma unroll
            for (int ms = 0; ms < 4; ++ms) {
                // A 16x32 bf16: lane m=0..15 (both halves), K segs hi*8 and 16+hi*8
                const __bf16* ap =
                    &As[(wm * 64 + ms * 16 + m15) * AS_STRIDE + kb * 32 + hi * 8];
                v8bf lo = *(const v8bf*)ap;
                v8bf h8 = *(const v8bf*)(ap + 16);
                afrag[ms] = __builtin_shufflevector(lo, h8, 0, 1, 2, 3, 4, 5, 6, 7,
                                                    8, 9, 10, 11, 12, 13, 14, 15);
            }
            v16bf bfrag[2];
#pragma unroll
            for (int nsub = 0; nsub < 2; ++nsub) {
                // B 32x16 bf16: lane n=0..15 (both halves), K segs hi*8 and 16+hi*8
                const __bf16* bp =
                    &Ws[(kb * 32 + hi * 8) * WS_STRIDE + wn * 32 + nsub * 16 + m15];
#pragma unroll
                for (int s = 0; s < 16; ++s)
                    bfrag[nsub][s] = bp[(s < 8 ? s : s + 8) * WS_STRIDE];
            }
#pragma unroll
            for (int ms = 0; ms < 4; ++ms)
#pragma unroll
                for (int nsub = 0; nsub < 2; ++nsub)
                    acc[ms][nsub] = __builtin_amdgcn_wmma_f32_16x16x32_bf16(
                        false, afrag[ms], false, bfrag[nsub], (short)0,
                        acc[ms][nsub], false, false);
        }
        __syncthreads();
    }

    // ---- epilogue: x = ns*(ss*y + sb) + nb  ->  a*y + b ----
#pragma unroll
    for (int nsub = 0; nsub < 2; ++nsub) {
        const int col = blockN + wn * 32 + nsub * 16 + m15;
        const float av = p_ns[col] * p_ss[col];
        const float bv = p_ns[col] * p_sb[col] + p_nb[col];
#pragma unroll
        for (int ms = 0; ms < 4; ++ms) {
            const int rbase = blockM + wm * 64 + ms * 16 + hi * 8;
#pragma unroll
            for (int r = 0; r < 8; ++r) {
                xout[(size_t)(rbase + r) * DDIM + col] = av * acc[ms][nsub][r] + bv;
            }
        }
    }
}

extern "C" void kernel_launch(void* const* d_in, const int* in_sizes, int n_in,
                              void* d_out, int out_size, void* d_ws, size_t ws_size,
                              hipStream_t stream) {
    const float* x           = (const float*)d_in[0];
    // d_in[1] = grids: unused (uniform grid is a compile-time constant)
    const float* coefs       = (const float*)d_in[2];   // [L, D, D, 8]
    const float* scale_base  = (const float*)d_in[3];   // [L, D, D]
    const float* scale_sp    = (const float*)d_in[4];   // [L, D, D]
    const float* sub_scale   = (const float*)d_in[5];   // [L, D]
    const float* sub_bias    = (const float*)d_in[6];
    const float* node_scale  = (const float*)d_in[7];
    const float* node_bias   = (const float*)d_in[8];
    float* out = (float*)d_out;

    // workspace layout
    __bf16* Wa = (__bf16*)d_ws;                                   // KTOT*DDIM bf16
    char* p = (char*)d_ws + (size_t)KTOT * DDIM * sizeof(__bf16);
    float* x1 = (float*)p;
    float* x2 = x1 + (size_t)BATCH * DDIM;

    const dim3 gemm_grid(DDIM / BN, BATCH / BM);   // 4 x 64
    const int pack_total = KTOT * DDIM;

    for (int l = 0; l < 3; ++l) {
        const float* xin = (l == 0) ? x : ((l == 1) ? x1 : x2);
        float* xout      = (l == 0) ? x1 : ((l == 1) ? x2 : out);

        kan_pack_kernel<<<pack_total / 256, 256, 0, stream>>>(
            coefs + (size_t)l * DDIM * DDIM * 8,
            scale_base + (size_t)l * DDIM * DDIM,
            scale_sp + (size_t)l * DDIM * DDIM, Wa);

        kan_layer_kernel<<<gemm_grid, THREADS, 0, stream>>>(
            xin, Wa,
            sub_scale + l * DDIM, sub_bias + l * DDIM,
            node_scale + l * DDIM, node_bias + l * DDIM, xout);
    }
}
